// ToeplitzCausalLinear_8400956031055
// MI455X (gfx1250) — matrix operand order repaired and checked by reading, not compiled, not run on hardware
//
#include <hip/hip_runtime.h>

// ToeplitzCausalLinear on MI455X (gfx1250):
//   out[m, t] = sum_{s<=t} x[m, s] * weight[t - s] + bias[t]
// GEMM (M=4096 x K=2048) x (K=2048 x N=2048), upper-triangular Toeplitz B.
// bf16 WMMA (V_WMMA_F32_16X16X32_BF16), f32 accumulate.
//
// B is never staged per K-step: a reversed zero-padded bf16 weight array R
// (8 shift-staggered LDS copies for 16B alignment) makes every B fragment a
// ds_load_b128 pair at R[i], R[i+16]. A tile is double-buffered (1 barrier/step).

#define DIM     2048
#define TILE_M  128
#define TILE_N  128
#define TILE_K  32
#define AROW_B  80      // A-tile row stride: 32 bf16 (64B) + 16B pad
#define SLEN    2176    // valid elements per reversed-weight copy (covers i<=2174)
#define SSTRIDE 2192    // padded stride between the 8 shifted copies (bank stagger)

typedef __attribute__((ext_vector_type(16))) __bf16       v16bf;
typedef __attribute__((ext_vector_type(8)))  float        v8f;
typedef __attribute__((ext_vector_type(4)))  unsigned int v4u;

union Frag { v16bf v; v4u q[2]; };

__device__ __forceinline__ unsigned short f2bf(float f) {
    union { __bf16 h; unsigned short u; } c;
    c.h = (__bf16)f;
    return c.u;
}
__device__ __forceinline__ unsigned bf16pk(float a, float b) {
    union { __bf16 h[2]; unsigned u; } c;
    c.h[0] = (__bf16)a;
    c.h[1] = (__bf16)b;
    return c.u;
}

__global__ __launch_bounds__(256)
void toeplitz_wmma_bf16(const float* __restrict__ x,
                        const float* __restrict__ weight,
                        const float* __restrict__ bias,
                        float* __restrict__ out) {
    // 8 shifted copies of reversed weight, bf16 bits: Sp[e] = R[e+p],
    // R[i] = (i < DIM) ? w[DIM-1-i] : 0.   (~34.3 KB)
    __shared__ unsigned short Rs[8 * SSTRIDE];
    // Double-buffered A tile, bf16 (2 x 10 KB).
    __shared__ unsigned int   AsU[2][TILE_M * (AROW_B / 4)];

    const int t    = threadIdx.x;
    const int n0   = blockIdx.x * TILE_N;
    const int m0   = blockIdx.y * TILE_M;
    const int wv   = t >> 5;        // wave id 0..7 -> 16-row M strip
    const int lane = t & 31;
    const int half = lane >> 4;
    const int l16  = lane & 15;

    // ---- Stage A tile for k-step `ks` into buffer `buf` ----
    auto stageA = [&](int buf, int k0) {
#pragma unroll
        for (int ii = 0; ii < 4; ++ii) {
            const int idx = t + ii * 256;           // 1024 float4 groups
            const int row = idx >> 3;
            const int c4  = idx & 7;
            const float4 f =
                *(const float4*)(x + (size_t)(m0 + row) * DIM + k0 + c4 * 4);
            uint2 p;
            p.x = bf16pk(f.x, f.y);
            p.y = bf16pk(f.z, f.w);
            *(uint2*)((char*)AsU[buf] + row * AROW_B + c4 * 8) = p;
        }
    };

    stageA(0, 0);

    // ---- Build the 8 shifted reversed-weight copies (once per block) ----
    for (int idx = t; idx < 8 * SSTRIDE; idx += 256) {
        const int p = idx / SSTRIDE;
        const int e = idx - p * SSTRIDE;
        const int i = e + p;                        // index into conceptual R
        const float v = (i < DIM) ? weight[DIM - 1 - i] : 0.f;
        Rs[idx] = f2bf(v);
    }

    v8f acc[8] = {};

    // Causality: only K-tiles with k0 < n0 + TILE_N contribute.
    int kmax = n0 + TILE_N;
    if (kmax > DIM) kmax = DIM;
    const int nk = kmax / TILE_K;

    // Per-lane B-fragment base index into R (before k0 / nt terms):
    // i = DIM-1 - (n0 + nt*16 + l16) + (k0 + half*8)
    const int ibase0 = DIM - 1 - n0 - l16 + half * 8;

    for (int ks = 0; ks < nk; ++ks) {
        __syncthreads();   // tile `ks` ready; buffer (ks+1)&1 free to overwrite
        if (ks + 1 < nk) stageA((ks + 1) & 1, (ks + 1) * TILE_K);

        const int k0 = ks * TILE_K;

        // A fragment (16x32 bf16), ISA layout: lane L: m=L&15, K-base=(L>>4)*8.
        Frag a;
        {
            const char* ap =
                (const char*)AsU[ks & 1] + (wv * 16 + l16) * AROW_B + half * 16;
            a.q[0] = *(const v4u*)ap;          // K = kb .. kb+7
            a.q[1] = *(const v4u*)(ap + 32);   // K = 16+kb .. 16+kb+7
        }

#pragma unroll
        for (int nt = 0; nt < 8; ++nt) {
            // B fragment = R[i..i+7] then R[i+16..i+23] (i always >= 0);
            // pick shifted copy p=i&7 so the load is 16B aligned.
            const int i = ibase0 + k0 - nt * 16;
            const int p = i & 7;
            const int e = i - p;               // multiple of 8 -> 16B aligned
            const char* bp = (const char*)Rs + (size_t)(p * SSTRIDE + e) * 2;
            Frag b;
            b.q[0] = *(const v4u*)bp;
            b.q[1] = *(const v4u*)(bp + 32);
            acc[nt] = __builtin_amdgcn_wmma_f32_16x16x32_bf16(
                false, a.v, false, b.v, (short)0, acc[nt], false, false);
        }
    }

    // ---- Epilogue: C layout VGPR r -> row half*8+r, col = l16; fuse bias.
    // Non-temporal stores: out is write-once, keep x resident in L2.
    const int rowbase = m0 + wv * 16 + half * 8;
#pragma unroll
    for (int nt = 0; nt < 8; ++nt) {
        const int col = n0 + nt * 16 + l16;
        const float bv = bias[col];
#pragma unroll
        for (int r = 0; r < 8; ++r)
            __builtin_nontemporal_store(acc[nt][r] + bv,
                                        out + (size_t)(rowbase + r) * DIM + col);
    }
}

extern "C" void kernel_launch(void* const* d_in, const int* in_sizes, int n_in,
                              void* d_out, int out_size, void* d_ws, size_t ws_size,
                              hipStream_t stream) {
    const float* x      = (const float*)d_in[0];  // (B, E, S) fp32
    const float* weight = (const float*)d_in[1];  // (1, S)    fp32
    const float* bias   = (const float*)d_in[2];  // (S,)      fp32
    float*       out    = (float*)d_out;          // (B, E, S) fp32

    const int M = in_sizes[0] / DIM;              // 4096
    dim3 grid(DIM / TILE_N, M / TILE_M);          // (16, 32)
    toeplitz_wmma_bf16<<<grid, 256, 0, stream>>>(x, weight, bias, out);
}
